// LinearTransformer_77884936946293
// MI455X (gfx1250) — compile-verified, matrix operand order checked by reading, not run on hardware
//
#include <hip/hip_runtime.h>

// -------- problem dims (fixed by reference) --------
static constexpr int BLK = 8;       // transformer blocks
static constexpr int Hh  = 16;      // heads
static constexpr int Dd  = 1024;    // model dim
static constexpr int KH  = 64;      // head dim
static constexpr int Ff  = 4096;    // ffn dim
static constexpr int NB  = 4;       // batch
static constexpr int Ss  = 4096;    // seq
static constexpr int HK  = Hh * KH; // 1024
static constexpr int Mm  = NB * Ss; // 16384 rows of activations

typedef __bf16 bf16_t;
typedef __bf16 v16bf __attribute__((ext_vector_type(16)));
typedef __bf16 v8bf  __attribute__((ext_vector_type(8)));
typedef float  v8f   __attribute__((ext_vector_type(8)));

enum { OP_NONE = 0, OP_GELU_F32 = 1, OP_GELU_BF16 = 2, OP_ELU1 = 3 };

// CDNA5 direct global->LDS async copy path (ASYNCcnt), if the toolchain has it.
#if defined(__gfx1250__) && \
    __has_builtin(__builtin_amdgcn_global_load_async_to_lds_b128) && \
    __has_builtin(__builtin_amdgcn_s_wait_asynccnt)
#define USE_ASYNC_LDS 1
#else
#define USE_ASYNC_LDS 0
#endif

// -------- math helpers --------
__device__ __forceinline__ float gelu_f(float x) {
    // tanh approximation (jax.nn.gelu default)
    float x3 = x * x * x;
    return 0.5f * x * (1.0f + tanhf(0.79788456080286535588f * (x + 0.044715f * x3)));
}
__device__ __forceinline__ float elu1_f(float x) {
    // elu(x) + 1  ==  x>0 ? x+1 : exp(x)
    return x > 0.0f ? x + 1.0f : __expf(x);
}

// =====================================================================
// Tiled bf16 WMMA GEMM:  C[M,N] = op(A[M,K] * Bt[N,K]^T + bias[N])
//   A  : bf16 row-major [M,K]
//   Bt : bf16 row-major [N,K]   (weights pre-transposed at conversion)
// Block tile 128x128x32, 256 threads = 8 waves (4 in M x 2 in N),
// each wave computes 32x64 = 2x4 WMMA 16x16 tiles.
// gfx1250 staging: GLOBAL_LOAD_ASYNC_TO_LDS_B128 (memory->LDS direct,
// tracked by ASYNCcnt), triple-buffered with TWO tiles in flight;
// steady state waits asynccnt<=4 (4 copies/wave per tile) so only the
// oldest tile is retired while the newer copy keeps streaming.
// =====================================================================
#define BM 128
#define BN 128
#define BK 32
#define LDK 40   // padded LDS row (bf16 elems): 80B rows, 16B-aligned chunks
#define NSTAGE 3

template <int OP>
__global__ __launch_bounds__(256)
void gemm_bf16_wmma(const bf16_t* __restrict__ A,
                    const bf16_t* __restrict__ Bt,
                    const float*  __restrict__ bias,
                    float*        __restrict__ Cf,
                    bf16_t*       __restrict__ Ch,
                    int M, int N, int Kd)
{
    __shared__ __align__(16) bf16_t As[NSTAGE][BM * LDK];
    __shared__ __align__(16) bf16_t Bs[NSTAGE][BN * LDK];

    const int tid  = threadIdx.x;
    const int wave = tid >> 5;        // 0..7
    const int lane = tid & 31;
    const int wm   = wave >> 1;       // 0..3  (M direction)
    const int wn   = wave & 1;        // 0..1  (N direction)
    const int lh   = lane >> 4;       // K-half select (WMMA 16-bit operand layout)
    const int lr   = lane & 15;       // row/col within 16

    const int rowBase = blockIdx.y * BM;
    const int colBase = blockIdx.x * BN;

    // staging: 512 chunks of 8 bf16 (16B) per tile, 2 chunks per thread
    const int c0  = tid * 2;
    const int ar0 = c0 >> 2,       ac0 = (c0 & 3) * 8;
    const int ar1 = (c0 + 1) >> 2, ac1 = ((c0 + 1) & 3) * 8;

    const bf16_t* gA0 = A  + (size_t)(rowBase + ar0) * Kd + ac0;
    const bf16_t* gA1 = A  + (size_t)(rowBase + ar1) * Kd + ac1;
    const bf16_t* gB0 = Bt + (size_t)(colBase + ar0) * Kd + ac0;
    const bf16_t* gB1 = Bt + (size_t)(colBase + ar1) * Kd + ac1;

    v8f acc[2][4] = {};

    auto compute = [&](int buf) {
        // Fragment loads per documented 16-bit operand striping:
        // lanes 0-15 hold K {0..7,16..23}, lanes 16-31 hold K {8..15,24..31}.
        v16bf af[2], bfm[4];
        #pragma unroll
        for (int mt = 0; mt < 2; ++mt) {
            const bf16_t* p = &As[buf][(wm * 32 + mt * 16 + lr) * LDK];
            v8bf lo = *(const v8bf*)(p + lh * 8);
            v8bf hi = *(const v8bf*)(p + 16 + lh * 8);
            af[mt] = __builtin_shufflevector(lo, hi,
                     0,1,2,3,4,5,6,7,8,9,10,11,12,13,14,15);
        }
        #pragma unroll
        for (int nt = 0; nt < 4; ++nt) {
            const bf16_t* p = &Bs[buf][(wn * 64 + nt * 16 + lr) * LDK];
            v8bf lo = *(const v8bf*)(p + lh * 8);
            v8bf hi = *(const v8bf*)(p + 16 + lh * 8);
            bfm[nt] = __builtin_shufflevector(lo, hi,
                      0,1,2,3,4,5,6,7,8,9,10,11,12,13,14,15);
        }
        #pragma unroll
        for (int mt = 0; mt < 2; ++mt)
            #pragma unroll
            for (int nt = 0; nt < 4; ++nt)
                acc[mt][nt] = __builtin_amdgcn_wmma_f32_16x16x32_bf16(
                    false, af[mt], false, bfm[nt],
                    (short)0, acc[mt][nt], false, false);
    };

    const int nk = Kd / BK;   // >= 32 for all call sites

#if USE_ASYNC_LDS
    // ---- async global->LDS pipeline, 2 tiles in flight (ASYNCcnt) ----
    // Builtin signature (from toolchain): (int4 AS1*, int4 AS3*, imm, imm)
    typedef int i4 __attribute__((vector_size(16)));
    typedef __attribute__((address_space(1))) i4* gp_t;
    typedef __attribute__((address_space(3))) i4* lp_t;
    auto stage_async = [&](int kt, int buf) {
        __builtin_amdgcn_global_load_async_to_lds_b128(
            (gp_t)(gA0 + kt), (lp_t)&As[buf][ar0 * LDK + ac0], 0, 0);
        __builtin_amdgcn_global_load_async_to_lds_b128(
            (gp_t)(gA1 + kt), (lp_t)&As[buf][ar1 * LDK + ac1], 0, 0);
        __builtin_amdgcn_global_load_async_to_lds_b128(
            (gp_t)(gB0 + kt), (lp_t)&Bs[buf][ar0 * LDK + ac0], 0, 0);
        __builtin_amdgcn_global_load_async_to_lds_b128(
            (gp_t)(gB1 + kt), (lp_t)&Bs[buf][ar1 * LDK + ac1], 0, 0);
    };

    stage_async(0, 0);
    stage_async(BK, 1);
    __builtin_amdgcn_s_wait_asynccnt(4);   // tile 0 landed; tile 1 in flight
    __syncthreads();
    for (int t = 1; t < nk - 1; ++t) {
        stage_async((t + 1) * BK, (t + 1) % NSTAGE);
        if (t + 3 < nk) {                  // L2 warm-up beyond staged tiles
            __builtin_prefetch(gA0 + (t + 3) * BK, 0, 3);
            __builtin_prefetch(gB0 + (t + 3) * BK, 0, 3);
        }
        compute((t - 1) % NSTAGE);
        __builtin_amdgcn_s_wait_asynccnt(4);  // retire oldest tile only
        __syncthreads();
    }
    compute((nk - 2) % NSTAGE);
    __builtin_amdgcn_s_wait_asynccnt(0);      // final tile complete
    __syncthreads();
    compute((nk - 1) % NSTAGE);
#else
    // ---- register-staged pipeline (fallback / host pass) ----
    uint4 ra0, ra1, rb0, rb1;
    auto loadg = [&](int kt) {
        ra0 = *(const uint4*)(gA0 + kt);
        ra1 = *(const uint4*)(gA1 + kt);
        rb0 = *(const uint4*)(gB0 + kt);
        rb1 = *(const uint4*)(gB1 + kt);
    };
    auto stores = [&](int buf) {
        *(uint4*)(&As[buf][ar0 * LDK + ac0]) = ra0;
        *(uint4*)(&As[buf][ar1 * LDK + ac1]) = ra1;
        *(uint4*)(&Bs[buf][ar0 * LDK + ac0]) = rb0;
        *(uint4*)(&Bs[buf][ar1 * LDK + ac1]) = rb1;
    };
    loadg(0);
    stores(0);
    __syncthreads();
    for (int t = 1; t < nk; ++t) {
        loadg(t * BK);
        if (t + 1 < nk) {
            __builtin_prefetch(gA0 + (t + 1) * BK, 0, 3);
            __builtin_prefetch(gB0 + (t + 1) * BK, 0, 3);
        }
        compute((t - 1) & 1);
        stores(t & 1);
        __syncthreads();
    }
    compute((nk - 1) & 1);
#endif

    // epilogue: C/D layout — VGPR r: lanes0-15 -> M=r, lanes16-31 -> M=8+r; N=lr
    const int mWave = rowBase + wm * 32;
    const int nWave = colBase + wn * 64;
    #pragma unroll
    for (int mt = 0; mt < 2; ++mt) {
        #pragma unroll
        for (int nt = 0; nt < 4; ++nt) {
            const int n  = nWave + nt * 16 + lr;
            const int m0 = mWave + mt * 16 + lh * 8;
            const float bcol = bias ? bias[n] : 0.0f;
            #pragma unroll
            for (int r = 0; r < 8; ++r) {
                float c = acc[mt][nt][r] + bcol;
                const size_t idx = (size_t)(m0 + r) * N + n;
                if constexpr (OP == OP_GELU_BF16) {
                    Ch[idx] = (bf16_t)gelu_f(c);
                } else {
                    if constexpr (OP == OP_GELU_F32) c = gelu_f(c);
                    if constexpr (OP == OP_ELU1)     c = elu1_f(c);
                    Cf[idx] = c;
                }
            }
        }
    }
}

// =====================================================================
// Weight conversion kernels (fp32 -> bf16, pre-transposed for the GEMM)
// =====================================================================

// Wq/Wk/Wv: [B][H][D][K] -> per-block [H*K][D]
__global__ void cvt_qkvT(const float* __restrict__ W, bf16_t* __restrict__ out) {
    const size_t i = (size_t)blockIdx.x * blockDim.x + threadIdx.x; // over B*H*D*K
    const size_t k   = i & 63;
    const size_t d   = (i >> 6) & 1023;
    const size_t h   = (i >> 16) & 15;
    const size_t blk = i >> 20;
    out[(blk << 20) + (h * 64 + k) * 1024 + d] = (bf16_t)W[i];
}

// generic per-block transpose: [B][R][C] -> [B][C][R]
__global__ void cvt_T(const float* __restrict__ W, bf16_t* __restrict__ out,
                      int R, int C) {
    const size_t i  = (size_t)blockIdx.x * blockDim.x + threadIdx.x; // over B*R*C
    const size_t rc = (size_t)R * C;
    const size_t blk = i / rc;
    const size_t rem = i - blk * rc;
    const size_t r = rem / C, c = rem - r * C;
    out[blk * rc + c * (size_t)R + r] = (bf16_t)W[i];
}

// x -> working fp32 copy + bf16 mirror
__global__ void cvt_x(const float* __restrict__ x, float* __restrict__ xf,
                      bf16_t* __restrict__ xh) {
    const size_t i = (size_t)blockIdx.x * blockDim.x + threadIdx.x;
    const float v = x[i];
    xf[i] = v;
    xh[i] = (bf16_t)v;
}

// =====================================================================
// Linear-attention pieces
// =====================================================================

// stage 1: partial sums over S chunks. grid (16 schunks, 4 batch, 4 col-chunks)
__global__ void kv_partial(const float* __restrict__ Kf, const float* __restrict__ V,
                           float* __restrict__ kvP, float* __restrict__ skP) {
    const int c  = blockIdx.z * blockDim.x + threadIdx.x;  // 0..1023
    const int n  = blockIdx.y;                             // 0..3
    const int sc = blockIdx.x;                             // 0..15
    const int SCH = Ss / 16;
    size_t base = ((size_t)n * Ss + (size_t)sc * SCH) * HK + c;
    float kv = 0.0f, sk = 0.0f;
    for (int s = 0; s < SCH; ++s) {
        const float k = Kf[base];
        const float v = V[base];
        kv += k * v;
        sk += k;
        base += HK;
    }
    const size_t o = ((size_t)n * 16 + sc) * HK + c;
    kvP[o] = kv;
    skP[o] = sk;
}

// stage 2: reduce the 16 partials -> KV[n][c], sumK[n][c]
__global__ void kv_final(const float* __restrict__ kvP, const float* __restrict__ skP,
                         float* __restrict__ KV, float* __restrict__ SK) {
    const int idx = blockIdx.x * blockDim.x + threadIdx.x; // 0..4095
    const int n = idx >> 10, c = idx & 1023;
    float kv = 0.0f, sk = 0.0f;
    for (int sc = 0; sc < 16; ++sc) {
        const size_t o = ((size_t)n * 16 + sc) * HK + c;
        kv += kvP[o];
        sk += skP[o];
    }
    KV[idx] = kv;
    SK[idx] = sk;
}

// att = Q * KV / (Q * sumK + 1e-6)  -> bf16 for the Wo GEMM
__global__ void att_kernel(const float* __restrict__ Q, const float* __restrict__ KV,
                           const float* __restrict__ SK, bf16_t* __restrict__ attb) {
    const size_t i = (size_t)blockIdx.x * blockDim.x + threadIdx.x;
    const int c = (int)(i & 1023);
    const int n = (int)(i >> 22);           // S*HK = 2^22
    const size_t nc = ((size_t)n << 10) + c;
    const float q = Q[i];
    attb[i] = (bf16_t)(q * KV[nc] / (q * SK[nc] + 1e-6f));
}

// fused residual + LayerNorm over D=1024 (one block per row), writes fp32 + bf16
__global__ __launch_bounds__(256)
void ln_residual(const float* __restrict__ Y, float* __restrict__ Xf,
                 bf16_t* __restrict__ Xh, const float* __restrict__ sc,
                 const float* __restrict__ bi) {
    __shared__ float red[18];
    const int row = blockIdx.x;
    const int tid = threadIdx.x;
    const size_t base = (size_t)row * Dd;

    float t[4], sum = 0.0f, sq = 0.0f;
    #pragma unroll
    for (int j = 0; j < 4; ++j) {
        const float v = Y[base + tid + j * 256] + Xf[base + tid + j * 256];
        t[j] = v;
        sum += v;
        sq  += v * v;
    }
    // wave32 reduction
    #pragma unroll
    for (int off = 16; off > 0; off >>= 1) {
        sum += __shfl_down(sum, off, 32);
        sq  += __shfl_down(sq,  off, 32);
    }
    if ((tid & 31) == 0) { red[tid >> 5] = sum; red[8 + (tid >> 5)] = sq; }
    __syncthreads();
    if (tid == 0) {
        float s0 = 0.0f, s1 = 0.0f;
        for (int w = 0; w < 8; ++w) { s0 += red[w]; s1 += red[8 + w]; }
        red[16] = s0; red[17] = s1;
    }
    __syncthreads();
    const float mean = red[16] * (1.0f / Dd);
    const float var  = red[17] * (1.0f / Dd) - mean * mean;
    const float inv  = rsqrtf(var + 1e-6f);
    #pragma unroll
    for (int j = 0; j < 4; ++j) {
        const int col = tid + j * 256;
        const float o = (t[j] - mean) * inv * sc[col] + bi[col];
        Xf[base + col] = o;
        Xh[base + col] = (bf16_t)o;
    }
}

// =====================================================================
// Host orchestration
// =====================================================================
extern "C" void kernel_launch(void* const* d_in, const int* in_sizes, int n_in,
                              void* d_out, int out_size, void* d_ws, size_t ws_size,
                              hipStream_t stream) {
    const float* x    = (const float*)d_in[0];
    const float* Wq   = (const float*)d_in[1];
    const float* bq   = (const float*)d_in[2];
    const float* Wk   = (const float*)d_in[3];
    const float* bk   = (const float*)d_in[4];
    const float* Wv   = (const float*)d_in[5];
    const float* bv   = (const float*)d_in[6];
    const float* Wo   = (const float*)d_in[7];
    const float* bo   = (const float*)d_in[8];
    const float* ln1s = (const float*)d_in[9];
    const float* ln1b = (const float*)d_in[10];
    const float* W1   = (const float*)d_in[11];
    const float* b1   = (const float*)d_in[12];
    const float* W2   = (const float*)d_in[13];
    const float* b2   = (const float*)d_in[14];
    const float* ln2s = (const float*)d_in[15];
    const float* ln2b = (const float*)d_in[16];

    char* ws = (char*)d_ws;
    size_t off = 0;
    auto take = [&](size_t bytes) -> char* {
        char* p = ws + off;
        off += (bytes + 255) & ~(size_t)255;
        return p;
    };

    const size_t DHK  = (size_t)Dd * HK;   // 1M elems
    const size_t DF   = (size_t)Dd * Ff;   // 4M elems
    const size_t MD   = (size_t)Mm * Dd;   // 16M elems
    const size_t MF   = (size_t)Mm * Ff;   // 64M elems

    bf16_t* wqT = (bf16_t*)take(BLK * DHK * sizeof(bf16_t));
    bf16_t* wkT = (bf16_t*)take(BLK * DHK * sizeof(bf16_t));
    bf16_t* wvT = (bf16_t*)take(BLK * DHK * sizeof(bf16_t));
    bf16_t* woT = (bf16_t*)take(BLK * DHK * sizeof(bf16_t));
    bf16_t* w1T = (bf16_t*)take(BLK * DF  * sizeof(bf16_t));
    bf16_t* w2T = (bf16_t*)take(BLK * DF  * sizeof(bf16_t));

    float*  xf   = (float*) take(MD * sizeof(float));
    bf16_t* xh   = (bf16_t*)take(MD * sizeof(bf16_t));
    float*  Qb   = (float*) take(MD * sizeof(float));
    float*  Kb   = (float*) take(MD * sizeof(float));
    float*  Vb   = (float*) take(MD * sizeof(float));
    bf16_t* attb = (bf16_t*)take(MD * sizeof(bf16_t));
    bf16_t* hb   = (bf16_t*)take(MF * sizeof(bf16_t));
    float*  yt   = (float*) take(MD * sizeof(float));
    float*  kvP  = (float*) take((size_t)NB * 16 * HK * sizeof(float));
    float*  skP  = (float*) take((size_t)NB * 16 * HK * sizeof(float));
    float*  KVb  = (float*) take((size_t)NB * HK * sizeof(float));
    float*  SKb  = (float*) take((size_t)NB * HK * sizeof(float));

    // ---- per-launch weight/activation conversions ----
    cvt_qkvT<<<(BLK * DHK) / 256, 256, 0, stream>>>(Wq, wqT);
    cvt_qkvT<<<(BLK * DHK) / 256, 256, 0, stream>>>(Wk, wkT);
    cvt_qkvT<<<(BLK * DHK) / 256, 256, 0, stream>>>(Wv, wvT);
    cvt_T   <<<(BLK * DHK) / 256, 256, 0, stream>>>(Wo, woT, HK, Dd);
    cvt_T   <<<(BLK * DF)  / 256, 256, 0, stream>>>(W1, w1T, Dd, Ff);
    cvt_T   <<<(BLK * DF)  / 256, 256, 0, stream>>>(W2, w2T, Ff, Dd);
    cvt_x   <<<MD / 256, 256, 0, stream>>>(x, xf, xh);

    const dim3 blk256(256);
    const dim3 gemmGridD(Dd / BN, Mm / BM);   // N=1024 outputs
    const dim3 gemmGridF(Ff / BN, Mm / BM);   // N=4096 outputs

    for (int i = 0; i < BLK; ++i) {
        const bf16_t* wq = wqT + (size_t)i * DHK;
        const bf16_t* wk = wkT + (size_t)i * DHK;
        const bf16_t* wv = wvT + (size_t)i * DHK;
        const bf16_t* wo = woT + (size_t)i * DHK;
        const bf16_t* w1 = w1T + (size_t)i * DF;
        const bf16_t* w2 = w2T + (size_t)i * DF;

        // Q/K with fused feature map (elu+1), V plain
        gemm_bf16_wmma<OP_ELU1><<<gemmGridD, blk256, 0, stream>>>(
            xh, wq, bq + (size_t)i * HK, Qb, nullptr, Mm, HK, Dd);
        gemm_bf16_wmma<OP_ELU1><<<gemmGridD, blk256, 0, stream>>>(
            xh, wk, bk + (size_t)i * HK, Kb, nullptr, Mm, HK, Dd);
        gemm_bf16_wmma<OP_NONE><<<gemmGridD, blk256, 0, stream>>>(
            xh, wv, bv + (size_t)i * HK, Vb, nullptr, Mm, HK, Dd);

        // linear attention normalizers
        kv_partial<<<dim3(16, NB, HK / 256), blk256, 0, stream>>>(Kb, Vb, kvP, skP);
        kv_final  <<<(NB * HK) / 256, blk256, 0, stream>>>(kvP, skP, KVb, SKb);
        att_kernel<<<MD / 256, blk256, 0, stream>>>(Qb, KVb, SKb, attb);

        // output projection + gelu, then residual+LN1
        gemm_bf16_wmma<OP_GELU_F32><<<gemmGridD, blk256, 0, stream>>>(
            attb, wo, bo + (size_t)i * Dd, yt, nullptr, Mm, Dd, HK);
        ln_residual<<<Mm, blk256, 0, stream>>>(
            yt, xf, xh, ln1s + (size_t)i * Dd, ln1b + (size_t)i * Dd);

        // FFN: gelu(x@W1+b1) -> bf16 h ; gelu(h@W2+b2) -> fp32 ; residual+LN2
        gemm_bf16_wmma<OP_GELU_BF16><<<gemmGridF, blk256, 0, stream>>>(
            xh, w1, b1 + (size_t)i * Ff, nullptr, hb, Mm, Ff, Dd);
        gemm_bf16_wmma<OP_GELU_F32><<<gemmGridD, blk256, 0, stream>>>(
            hb, w2, b2 + (size_t)i * Dd, yt, nullptr, Mm, Dd, Ff);
        ln_residual<<<Mm, blk256, 0, stream>>>(
            yt, xf, xh, ln2s + (size_t)i * Dd, ln2b + (size_t)i * Dd);
    }

    (void)hipMemcpyAsync(d_out, xf, (size_t)out_size * sizeof(float),
                         hipMemcpyDeviceToDevice, stream);
}
